// MultiHeadAttention_28887950033224
// MI455X (gfx1250) — matrix operand order repaired and checked by reading, not compiled, not run on hardware
//
#include <hip/hip_runtime.h>
#include <hip/hip_bf16.h>
#include <math.h>

typedef _Float16 half_t;
typedef __attribute__((ext_vector_type(16))) _Float16 v16h;
typedef __attribute__((ext_vector_type(8)))  _Float16 v8h;
typedef __attribute__((ext_vector_type(8)))  float    v8f;
typedef __attribute__((ext_vector_type(4)))  float    v4f;

constexpr int NB = 4, NL = 1024, ND = 1024, NH = 16, NDH = 64;
constexpr int BL = NB * NL;          // 4096 rows
constexpr int HD = NH * NDH;         // 1024

#define USE_ASYNC_PREV 1

// ---------------- x convert ----------------
__global__ void cvt_f32_to_f16(const float* __restrict__ src, half_t* __restrict__ dst, int n) {
  int i = blockIdx.x * blockDim.x + threadIdx.x;
  if (i < n) dst[i] = (half_t)src[i];
}

// ---------------- tiled transpose-convert: W (K x N f32) -> Wt (N x K f16) ----------------
__global__ __launch_bounds__(256) void transpose_w_tiled(const float* __restrict__ W,
                                                         half_t* __restrict__ Wt,
                                                         int K, int N) {
  __shared__ half_t tile[64 * 65];
  int tilesX = N >> 6;
  int tx = blockIdx.x % tilesX;        // n tile
  int ty = blockIdx.x / tilesX;        // k tile
  int n0 = tx * 64, k0 = ty * 64;
  for (int i = threadIdx.x; i < 4096; i += 256) {
    int r = i >> 6, c = i & 63;        // r: k offset, c: n offset (coalesced read)
    tile[r * 65 + c] = (half_t)W[(size_t)(k0 + r) * N + (n0 + c)];
  }
  __syncthreads();
  for (int i = threadIdx.x; i < 4096; i += 256) {
    int r = i >> 6, c = i & 63;        // r: n offset, c: k offset (coalesced write)
    Wt[(size_t)(n0 + r) * K + (k0 + c)] = tile[c * 65 + r];
  }
}

// ---------------- shared WMMA GEMM core: one wave, 16 rows x 64 cols, f16 inputs ----------------
__device__ __forceinline__ void gemm_core(const half_t* __restrict__ Arow,
                                          const half_t* __restrict__ Bt,
                                          int colBase, int K, int lhi, int lr,
                                          v8f acc[4]) {
  for (int k0 = 0; k0 < K; k0 += 32) {
    int kb = k0 + lhi * 16;
    v8h lo = *(const v8h*)(Arow + kb);
    v8h hi = *(const v8h*)(Arow + kb + 8);
    v16h a;
#pragma unroll
    for (int e = 0; e < 8; ++e) { a[e] = lo[e]; a[8 + e] = hi[e]; }
    int kbB = k0 + lhi * 16;
#pragma unroll
    for (int j = 0; j < 4; ++j) {
      v16h bf = *(const v16h*)(Bt + (size_t)(colBase + j * 16 + lr) * K + kbB);
      acc[j] = __builtin_amdgcn_wmma_f32_16x16x32_f16(false, a, false, bf,
                                                      (short)0, acc[j], false, false);
    }
  }
}

// ---------------- Q/K projection + bias + fused per-head LayerNorm -> f16 (B,H,L,DH) ----------------
__global__ __launch_bounds__(256) void gemm_qk_ln(const half_t* __restrict__ A,   // (B*L, D)
                                                  const half_t* __restrict__ Bt,  // (HD, D)
                                                  const float* __restrict__ bias, // HD
                                                  const float* __restrict__ g,    // DH
                                                  const float* __restrict__ beta, // DH
                                                  half_t* __restrict__ out) {     // (B,H,L,DH)
  const int wave = threadIdx.x >> 5, lane = threadIdx.x & 31;
  const int lr = lane & 15, lhi = lane >> 4;
  int job = blockIdx.x * 8 + wave;
  int rowTile = job >> 4;              // 16 strips per row tile (one head each)
  int h = job & 15;
  v8f acc[4];
#pragma unroll
  for (int j = 0; j < 4; ++j) acc[j] = {};
  const half_t* Arow = A + (size_t)(rowTile * 16 + lr) * ND;
  gemm_core(Arow, Bt, h * 64, ND, lhi, lr, acc);

  float gv[4], bev[4], bv[4];
#pragma unroll
  for (int j = 0; j < 4; ++j) {
    int d = j * 16 + lr;
    gv[j] = g[d]; bev[j] = beta[d]; bv[j] = bias[h * 64 + d];
  }
#pragma unroll
  for (int r = 0; r < 8; ++r) {
    float v0 = acc[0][r] + bv[0];
    float v1 = acc[1][r] + bv[1];
    float v2 = acc[2][r] + bv[2];
    float v3 = acc[3][r] + bv[3];
    float s = v0 + v1 + v2 + v3;
#pragma unroll
    for (int off = 8; off >= 1; off >>= 1) s += __shfl_xor(s, off, 32);  // over 16 lanes
    float m = s * (1.0f / 64.0f);
    float d0 = v0 - m, d1 = v1 - m, d2 = v2 - m, d3 = v3 - m;
    float q = d0 * d0 + d1 * d1 + d2 * d2 + d3 * d3;
#pragma unroll
    for (int off = 8; off >= 1; off >>= 1) q += __shfl_xor(q, off, 32);
    float inv = rsqrtf(q * (1.0f / 64.0f) + 1e-5f);
    int grow = rowTile * 16 + r + lhi * 8;
    int b = grow >> 10, l = grow & (NL - 1);
    half_t* dp = out + (((size_t)b * NH + h) * NL + l) * NDH;
    dp[0 + lr]  = (half_t)(d0 * inv * gv[0] + bev[0]);
    dp[16 + lr] = (half_t)(d1 * inv * gv[1] + bev[1]);
    dp[32 + lr] = (half_t)(d2 * inv * gv[2] + bev[2]);
    dp[48 + lr] = (half_t)(d3 * inv * gv[3] + bev[3]);
  }
}

// ---------------- V projection + bias, transposed f16 output (B,H,DH,L) via per-wave LDS ----------------
__global__ __launch_bounds__(256) void gemm_v_t(const half_t* __restrict__ A,
                                                const half_t* __restrict__ Bt,
                                                const float* __restrict__ bias,
                                                half_t* __restrict__ Vt) {
  __shared__ half_t vlds[8 * 16 * 64];
  const int wave = threadIdx.x >> 5, lane = threadIdx.x & 31;
  const int lr = lane & 15, lhi = lane >> 4;
  int job = blockIdx.x * 8 + wave;
  int rowTile = job >> 4;
  int h = job & 15;
  v8f acc[4];
#pragma unroll
  for (int j = 0; j < 4; ++j) acc[j] = {};
  const half_t* Arow = A + (size_t)(rowTile * 16 + lr) * ND;
  gemm_core(Arow, Bt, h * 64, ND, lhi, lr, acc);

  half_t* w = vlds + wave * (16 * 64);   // wave-private tile: [l_local][d]
#pragma unroll
  for (int j = 0; j < 4; ++j) {
    float bv = bias[h * 64 + j * 16 + lr];
#pragma unroll
    for (int r = 0; r < 8; ++r)
      w[(r + lhi * 8) * 64 + j * 16 + lr] = (half_t)(acc[j][r] + bv);
  }
  // wave-local transpose read (compiler inserts DScnt waits), 32B contiguous stores
  int b = (rowTile * 16) >> 10;
  int l0 = (rowTile * 16) & (NL - 1);
#pragma unroll
  for (int dd = 0; dd < 2; ++dd) {
    int d = 2 * lane + dd;               // conflict-free column gather
    v16h row;
#pragma unroll
    for (int i = 0; i < 16; ++i) row[i] = w[i * 64 + d];
    *(v16h*)(Vt + (((size_t)b * NH + h) * NDH + d) * NL + l0) = row;
  }
}

// ---------------- generic WMMA GEMM, fp32 output + bias (used for out-projection) ----------------
__global__ __launch_bounds__(256) void gemm_wmma(const half_t* __restrict__ A,
                                                 const half_t* __restrict__ Bt,
                                                 const float* __restrict__ bias,
                                                 float* __restrict__ C,
                                                 int M, int N, int K) {
  const int wave = threadIdx.x >> 5, lane = threadIdx.x & 31;
  const int lr = lane & 15, lhi = lane >> 4;
  const int nStrips = N >> 6;
  int job = blockIdx.x * 8 + wave;
  int rowTile = job / nStrips;
  int strip = job - rowTile * nStrips;
  if (rowTile * 16 >= M) return;
  v8f acc[4];
#pragma unroll
  for (int j = 0; j < 4; ++j) acc[j] = {};
  const half_t* Arow = A + (size_t)(rowTile * 16 + lr) * K;
  gemm_core(Arow, Bt, strip * 64, K, lhi, lr, acc);
#pragma unroll
  for (int j = 0; j < 4; ++j) {
    int col = strip * 64 + j * 16 + lr;
    float bv = bias ? bias[col] : 0.0f;
#pragma unroll
    for (int r = 0; r < 8; ++r) {
      int row = rowTile * 16 + r + lhi * 8;
      C[(size_t)row * N + col] = acc[j][r] + bv;
    }
  }
}

// ---------------- attention: one block per (b,h,16 q-rows) ----------------
// LDS: slds = 16x1024 f32 (64KB, prev tile -> scores -> exp -> phase3 staging),
//      wlds = 16x1024 f16 softmax weights (32KB).
__global__ __launch_bounds__(256) void attn_wmma(
    const half_t* __restrict__ Qh,   // (B,H,L,DH)
    const half_t* __restrict__ Kh,   // (B,H,L,DH)
    const half_t* __restrict__ Vt,   // (B,H,DH,L)
    const float* __restrict__ prev,  // (B,H,L,L)
    const unsigned char* __restrict__ kpm,  // (B,L)
    const float* __restrict__ scale_p,
    float* __restrict__ score_out,   // (B,H,L,L)
    float* __restrict__ weight_out,  // (B,H,L,L)
    half_t* __restrict__ embh) {     // (B,L,H*DH)
  extern __shared__ char smem[];
  float* slds = (float*)smem;
  half_t* wlds = (half_t*)(smem + 16 * NL * 4);

  const int wave = threadIdx.x >> 5, lane = threadIdx.x & 31;
  const int lr = lane & 15, lhi = lane >> 4;

  int bt = blockIdx.x;
  int qt = bt & 63;
  int bh = bt >> 6;
  int b = bh >> 4;
  int q0 = qt * 16;
  float scale = *scale_p;
  const float neg_inf = -__builtin_inff();
  size_t sbase = ((size_t)bh * NL + q0) * NL;

  // ---- Phase 0: async-stream this block's contiguous 64KB prev tile into LDS ----
  const char* prevBase = (const char*)(prev + sbase);
#if USE_ASYNC_PREV
  {
    unsigned ldsbase = (unsigned)(size_t)(void*)slds;
    for (int i = threadIdx.x; i < 4096; i += 256) {
      unsigned ldst = ldsbase + (unsigned)i * 16u;
      unsigned long long ga = (unsigned long long)(prevBase + (size_t)i * 16);
      asm volatile("global_load_async_to_lds_b128 %0, %1, off"
                   :: "v"(ldst), "v"(ga) : "memory");
    }
    asm volatile("s_wait_asynccnt 0x0" ::: "memory");
  }
#else
  for (int i = threadIdx.x; i < 4096; i += 256)
    ((v4f*)slds)[i] = ((const v4f*)prevBase)[i];
#endif

  // preload Q A-fragments (16 rows x 64 dims -> two K=32 fragments)
  const half_t* Qbase = Qh + ((size_t)bh * NL + q0) * NDH;
  v16h aq[2];
#pragma unroll
  for (int i = 0; i < 2; ++i) {
    int kb = i * 32 + lhi * 16;
    v8h lo = *(const v8h*)(Qbase + (size_t)lr * NDH + kb);
    v8h hi = *(const v8h*)(Qbase + (size_t)lr * NDH + kb + 8);
#pragma unroll
    for (int e = 0; e < 8; ++e) { aq[i][e] = lo[e]; aq[i][8 + e] = hi[e]; }
  }
  __syncthreads();  // prev tile resident

  // ---- Phase 1: S = QK^T*scale + prev(LDS), masked -> LDS in place ----
  for (int kt = wave; kt < 64; kt += 8) {
    v8f acc = {};
    const half_t* Kbase = Kh + ((size_t)bh * NL + kt * 16) * NDH;
#pragma unroll
    for (int i = 0; i < 2; ++i) {
      int kbB = i * 32 + lhi * 16;
      v16h bf = *(const v16h*)(Kbase + (size_t)lr * NDH + kbB);
      acc = __builtin_amdgcn_wmma_f32_16x16x32_f16(false, aq[i], false, bf,
                                                   (short)0, acc, false, false);
    }
    int key = kt * 16 + lr;
    bool pad = kpm[b * NL + key] != 0;
#pragma unroll
    for (int r = 0; r < 8; ++r) {
      int qloc = r + lhi * 8;
      int qrow = q0 + qloc;
      float s = acc[r] * scale + slds[qloc * NL + key];
      if (key > qrow || pad) s = neg_inf;
      slds[qloc * NL + key] = s;
    }
  }
  __syncthreads();

  // ---- Phase 1b: bulk-coalesced write of attn_score from LDS ----
  {
    v4f* dst = (v4f*)(score_out + sbase);
    const v4f* src = (const v4f*)slds;
    for (int i = threadIdx.x; i < 4096; i += 256) dst[i] = src[i];
  }
  __syncthreads();

  // ---- Phase 2: softmax rows (wave owns 2 rows); cache exp in LDS ----
#pragma unroll
  for (int rr = 0; rr < 2; ++rr) {
    int row = wave * 2 + rr;
    float m = neg_inf;
    for (int c = lane; c < NL; c += 32) m = fmaxf(m, slds[row * NL + c]);
#pragma unroll
    for (int off = 16; off >= 1; off >>= 1) m = fmaxf(m, __shfl_xor(m, off, 32));
    float sum = 0.0f;
    for (int c = lane; c < NL; c += 32) {
      float e = __expf(slds[row * NL + c] - m);
      slds[row * NL + c] = e;
      sum += e;
    }
#pragma unroll
    for (int off = 16; off >= 1; off >>= 1) sum += __shfl_xor(sum, off, 32);
    float inv = 1.0f / sum;
    size_t gb = sbase + (size_t)row * NL;
    for (int c = lane; c < NL; c += 32) {
      float w = slds[row * NL + c] * inv;
      weight_out[gb + c] = w;
      wlds[row * NL + c] = (half_t)w;
    }
  }
  __syncthreads();

  // ---- Phase 3: emb = W @ V (A from LDS, B from transposed V), split-K over wave pairs ----
  int dt = wave & 3;
  int khalf = wave >> 2;
  v8f acc = {};
  const half_t* Vb = Vt + ((size_t)bh * NDH + dt * 16) * (size_t)NL;
  for (int k0 = khalf * 512; k0 < khalf * 512 + 512; k0 += 32) {
    int kb = k0 + lhi * 16;
    v8h lo = *(const v8h*)(wlds + lr * NL + kb);
    v8h hi = *(const v8h*)(wlds + lr * NL + kb + 8);
    v16h a;
#pragma unroll
    for (int e = 0; e < 8; ++e) { a[e] = lo[e]; a[8 + e] = hi[e]; }
    int kbB = k0 + lhi * 16;
    v16h bf = *(const v16h*)(Vb + (size_t)lr * NL + kbB);
    acc = __builtin_amdgcn_wmma_f32_16x16x32_f16(false, a, false, bf,
                                                 (short)0, acc, false, false);
  }
  float* stage = slds;  // reuse
  if (khalf == 1) {
#pragma unroll
    for (int r = 0; r < 8; ++r) stage[dt * 256 + (r + lhi * 8) * 16 + lr] = acc[r];
  }
  __syncthreads();
  if (khalf == 0) {
#pragma unroll
    for (int r = 0; r < 8; ++r) {
      int qloc = r + lhi * 8;
      float v = acc[r] + stage[dt * 256 + qloc * 16 + lr];
      embh[((size_t)b * NL + (q0 + qloc)) * HD + (bh & (NH - 1)) * NDH + dt * 16 + lr] = (half_t)v;
    }
  }
}

extern "C" void kernel_launch(void* const* d_in, const int* in_sizes, int n_in,
                              void* d_out, int out_size, void* d_ws, size_t ws_size,
                              hipStream_t stream) {
  (void)in_sizes; (void)n_in; (void)out_size; (void)ws_size;
  const float* x    = (const float*)d_in[0];
  const float* prev = (const float*)d_in[1];
  // d_in[2] = attn_mask (causal; computed analytically in-kernel)
  const unsigned char* kpm = (const unsigned char*)d_in[3];
  const float* W_q = (const float*)d_in[4];
  const float* b_q = (const float*)d_in[5];
  const float* W_k = (const float*)d_in[6];
  const float* b_k = (const float*)d_in[7];
  const float* W_v = (const float*)d_in[8];
  const float* b_v = (const float*)d_in[9];
  const float* g_q = (const float*)d_in[10];
  const float* beta_q = (const float*)d_in[11];
  const float* g_k = (const float*)d_in[12];
  const float* beta_k = (const float*)d_in[13];
  const float* scale  = (const float*)d_in[14];
  const float* W_o = (const float*)d_in[15];
  const float* b_o = (const float*)d_in[16];

  char* ws = (char*)d_ws;
  size_t off = 0;
  auto alloc = [&](size_t bytes) -> char* {
    char* p = ws + off;
    off = (off + bytes + 255) & ~(size_t)255;
    return p;
  };
  half_t* xh   = (half_t*)alloc((size_t)BL * ND * 2);
  half_t* Wqt  = (half_t*)alloc((size_t)ND * HD * 2);
  half_t* Wkt  = (half_t*)alloc((size_t)ND * HD * 2);
  half_t* Wvt  = (half_t*)alloc((size_t)ND * HD * 2);
  half_t* Wot  = (half_t*)alloc((size_t)HD * ND * 2);
  half_t* Qh   = (half_t*)alloc((size_t)BL * HD * 2);
  half_t* Kh   = (half_t*)alloc((size_t)BL * HD * 2);
  half_t* Vth  = (half_t*)alloc((size_t)BL * HD * 2);
  half_t* embh = (half_t*)alloc((size_t)BL * HD * 2);

  float* emb_out    = (float*)d_out;
  float* weight_out = emb_out + (size_t)NB * NL * ND;
  float* score_out  = weight_out + (size_t)NB * NH * NL * NL;

  const int nx = BL * ND;
  cvt_f32_to_f16<<<(nx + 255) / 256, 256, 0, stream>>>(x, xh, nx);
  transpose_w_tiled<<<256, 256, 0, stream>>>(W_q, Wqt, ND, HD);
  transpose_w_tiled<<<256, 256, 0, stream>>>(W_k, Wkt, ND, HD);
  transpose_w_tiled<<<256, 256, 0, stream>>>(W_v, Wvt, ND, HD);
  transpose_w_tiled<<<256, 256, 0, stream>>>(W_o, Wot, HD, ND);

  // fused projections: 4096 wave-jobs -> 512 blocks each
  gemm_qk_ln<<<512, 256, 0, stream>>>(xh, Wqt, b_q, g_q, beta_q, Qh);
  gemm_qk_ln<<<512, 256, 0, stream>>>(xh, Wkt, b_k, g_k, beta_k, Kh);
  gemm_v_t<<<512, 256, 0, stream>>>(xh, Wvt, b_v, Vth);

  // attention: B*H*(L/16) = 4096 blocks, 96KB dynamic LDS
  attn_wmma<<<4096, 256, 96 * 1024, stream>>>(Qh, Kh, Vth, prev, kpm, scale,
                                              score_out, weight_out, embh);

  // output projection straight into emb region of d_out
  gemm_wmma<<<512, 256, 0, stream>>>(embh, Wot, b_o, emb_out, BL, ND, HD);
}
